// DyConvBlock_46677704573371
// MI455X (gfx1250) — matrix operand order repaired and checked by reading, not compiled, not run on hardware
//
#include <hip/hip_runtime.h>

// ---------------------------------------------------------------------------
// DyConvBlock for MI455X (gfx1250): dynamic-filter conv as implicit GEMM on
// v_wmma_f32_16x16x32_f16. Per-block, the per-sample filter bank (WMMA-A
// layout) is staged global->LDS with GLOBAL_LOAD_ASYNC_TO_LDS_B128
// (ASYNCcnt), overlapped with the f32->f16 x-tile fill. Deterministic
// two-stage reductions for SE pooling and GroupNorm stats; GN+LeakyReLU
// runs in place on d_out.
// ---------------------------------------------------------------------------

typedef __attribute__((ext_vector_type(16))) _Float16 v16h;
typedef __attribute__((ext_vector_type(8)))  float    v8f;

#define NB      16      // batch
#define CIN     64
#define COUT    64
#define HH      192
#define WW      192
#define HIDC    17
#define KDY     4
#define NGRP    8
#define PLANE   (HH*WW)             // 36864
#define TEMP_I  (1.0f/30.0f)
#define EPS_GN  1e-5f
#define NEG_SL  0.01f

#define TILES_W 12                  // 12 * 16 = 192
#define TILES_H 24                  // 24 * 8  = 192
#define NTILES  (TILES_W*TILES_H)   // 288 tiles per sample
#define NCHUNK  18                  // K = 576 = 18 chunks of 32 (tap-major)
#define A_HALVES (4*NCHUNK*32*16)   // per-sample A bank: 36864 halves = 72 KB

// ---- async global->LDS helpers (gfx1250 ASYNCcnt path) --------------------
#if __has_builtin(__builtin_amdgcn_global_load_async_to_lds_b128)
#define HAS_ASYNC_LDS 1
typedef int v4i __attribute__((vector_size(4 * sizeof(int))));
typedef __attribute__((address_space(1))) v4i* as1_v4p;   // global int4*
typedef __attribute__((address_space(3))) v4i* as3_v4p;   // LDS int4*
__device__ __forceinline__ void async_g2l_b128(const void* g, void* l) {
  // Low 32 bits of a generic pointer to __shared__ are the LDS byte offset.
  __builtin_amdgcn_global_load_async_to_lds_b128(
      (as1_v4p)(uintptr_t)g, (as3_v4p)(uintptr_t)(uint32_t)(uintptr_t)l,
      /*offset=*/0, /*cpol=*/0);
}
__device__ __forceinline__ void async_wait0() {
#if __has_builtin(__builtin_amdgcn_s_wait_asynccnt)
  __builtin_amdgcn_s_wait_asynccnt(0);
#else
  asm volatile("s_wait_asynccnt 0" ::: "memory");
#endif
}
#else
#define HAS_ASYNC_LDS 0
#endif

// ---------------------------------------------------------------------------
// K1: per-(b,c) partial sums of x for global average pooling (deterministic
//     block tree reduce; 9 blocks of 4096 elements per 36864-element plane).
// ---------------------------------------------------------------------------
__global__ __launch_bounds__(256) void pool_kernel(const float* __restrict__ x,
                                                   float* __restrict__ partials) {
  const int bc  = blockIdx.y;            // b*64 + c  (0..1023)
  const int blk = blockIdx.x;            // 0..8
  const int tid = threadIdx.x;
  const float* p = x + (size_t)bc * PLANE + blk * 4096;
  float s = 0.f;
#pragma unroll
  for (int i = 0; i < 16; ++i) s += p[i * 256 + tid];
  __shared__ float red[256];
  red[tid] = s;
  __syncthreads();
  for (int off = 128; off > 0; off >>= 1) {
    if (tid < off) red[tid] += red[tid + off];
    __syncthreads();
  }
  if (tid == 0) partials[bc * 9 + blk] = red[0];
}

// ---------------------------------------------------------------------------
// K2a: finish pooling, SE MLP (64 -> 17 relu -> 4), softmax(scores/T).
// ---------------------------------------------------------------------------
__global__ __launch_bounds__(256) void attn_kernel(const float* __restrict__ partials,
                                                   const float* __restrict__ w1,
                                                   const float* __restrict__ w2,
                                                   const float* __restrict__ b2,
                                                   float* __restrict__ attn) {
  __shared__ float pooled[NB * CIN];
  const int tid = threadIdx.x;
  for (int i = tid; i < NB * CIN; i += 256) {
    float s = 0.f;
    for (int j = 0; j < 9; ++j) s += partials[i * 9 + j];
    pooled[i] = s * (1.0f / (float)PLANE);
  }
  __syncthreads();
  if (tid < NB) {
    const int b = tid;
    float h[HIDC];
    for (int j = 0; j < HIDC; ++j) {
      float s = 0.f;
      for (int i = 0; i < CIN; ++i) s += pooled[b * CIN + i] * w1[j * CIN + i];
      h[j] = s > 0.f ? s : 0.f;
    }
    float sc[KDY], m = -1e30f;
    for (int k = 0; k < KDY; ++k) {
      float s = b2[k];
      for (int j = 0; j < HIDC; ++j) s += h[j] * w2[k * HIDC + j];
      sc[k] = s * TEMP_I;
      m = fmaxf(m, sc[k]);
    }
    float e[KDY], tot = 0.f;
    for (int k = 0; k < KDY; ++k) { e[k] = expf(sc[k] - m); tot += e[k]; }
    for (int k = 0; k < KDY; ++k) attn[b * KDY + k] = e[k] / tot;
  }
}

// ---------------------------------------------------------------------------
// K2b: synthesize per-sample f16 filters directly in the WMMA A-fragment
//      layout (ISA 7.12.2, 16-bit A 16x32):
//        lanes 0-15: M=lane, VGPR half j -> K = (j&7) + ((j>>3)<<4)
//        lanes16-31: same M rows, K offset +8
//      Global K ordering: kidx = tap*64 + ci, tap = kh*3+kw (chunk = 32 K's).
//      Buffer index: [b][t(4 m-tiles)][chunk(18)][lane(32)][j(16)] halves.
// ---------------------------------------------------------------------------
__global__ __launch_bounds__(256) void filt_kernel(const float* __restrict__ attn,
                                                   const float* __restrict__ wts,
                                                   _Float16* __restrict__ afilt) {
  const int idx = blockIdx.x * 256 + threadIdx.x;   // < 16*4*18*32*16 = 589824
  const int j    = idx & 15;
  const int L    = (idx >> 4) & 31;
  const int rest = idx >> 9;
  const int c    = rest % NCHUNK;
  const int tb   = rest / NCHUNK;
  const int t    = tb & 3;
  const int b    = tb >> 2;

  const int o  = t * 16 + (L & 15);
  const int kl = (j & 7) + ((j >> 3) << 4) + ((L >> 4) << 3);
  const int kg = c * 32 + kl;
  const int tap = kg >> 6;
  const int ci  = kg & 63;
  const int kh = tap / 3, kw = tap % 3;

  float v = 0.f;
#pragma unroll
  for (int k = 0; k < KDY; ++k)
    v += attn[b * KDY + k] * wts[((k * COUT + o) * CIN + ci) * 9 + kh * 3 + kw];
  afilt[idx] = (_Float16)v;
}

// ---------------------------------------------------------------------------
// K3: WMMA implicit-GEMM conv. One block = one sample, one 16(w) x 8(h) tile.
//     8 waves, one output row each; all 64 channels via 4 m-tiles.
//     Filter bank staged global->LDS via async B128 (overlapped with x fill);
//     x tile held in LDS as f16 with zero-filled halo. Per-(tile,group) GN
//     partials via deterministic shfl_xor trees.
// ---------------------------------------------------------------------------
__global__ __launch_bounds__(256) void conv_kernel(const float* __restrict__ x,
                                                   const _Float16* __restrict__ afilt,
                                                   float* __restrict__ y,
                                                   float* __restrict__ statp) {
  const int b   = blockIdx.z;
  const int thy = blockIdx.y;          // h tile (0..23)
  const int twx = blockIdx.x;          // w tile (0..11)
  const int h0 = thy * 8, w0 = twx * 16;
  const int tid = threadIdx.x, lane = tid & 31, wv = tid >> 5;

  __shared__ _Float16 xs[CIN * 10 * 18];   // x tile [ci][10][18], 23 KB
  __shared__ _Float16 as_[A_HALVES];       // filter bank [t][c][lane][j], 72 KB

  // Kick off async staging of this sample's filter bank (ASYNCcnt),
  // 4608 x B128 across 256 threads = 18 per thread.
  const _Float16* agsrc = afilt + (size_t)b * A_HALVES;
#if HAS_ASYNC_LDS
#pragma unroll
  for (int i = 0; i < A_HALVES / 8 / 256; ++i) {   // 18 iterations
    const int e8 = i * 256 + tid;                  // index in 8-half units
    async_g2l_b128(agsrc + e8 * 8, &as_[e8 * 8]);
  }
#else
  for (int e8 = tid; e8 < A_HALVES / 8; e8 += 256)
    *(float4*)&as_[e8 * 8] = *(const float4*)(agsrc + e8 * 8);
#endif

  // Overlap: fill + convert the x tile (zero-filled halo for pad=1).
  for (int i = tid; i < CIN * 10 * 18; i += 256) {
    const int ci = i / 180, r = i % 180, hl = r / 18, wl = r % 18;
    const int gh = h0 - 1 + hl, gw = w0 - 1 + wl;
    float v = 0.f;
    if ((unsigned)gh < (unsigned)HH && (unsigned)gw < (unsigned)WW)
      v = x[(((size_t)b * CIN + ci) * HH + gh) * WW + gw];
    xs[i] = (_Float16)v;
  }
#if HAS_ASYNC_LDS
  async_wait0();
#endif
  __syncthreads();

  const int hloc = wv;                 // this wave's output row within the tile
  v8f acc[4] = {};
  const v16h* aL = (const v16h*)as_;
  union { v16h v; _Float16 h[16]; } bf;

  for (int c = 0; c < NCHUNK; ++c) {
    const int tap = c >> 1;
    const int kh = tap / 3, kw = tap % 3;
    const int ci = ((c & 1) << 5);                   // B layout: N = lane&15, K_local = (j&7)+((j>>3)<<4)+8*(lane>>4)
    const _Float16* row = &xs[(ci * 10 + hloc + kh) * 18 + kw + (lane & 15)];
#pragma unroll
    for (int j = 0; j < 16; ++j) bf.h[j] = row[((j & 7) + ((j >> 3) << 4) + ((lane >> 4) << 3)) * 180];

    // Issue all four A-fragment LDS loads before the WMMAs so the scheduler
    // can batch ds_load_b128s under one wait.
    const v16h a0 = aL[(0 * NCHUNK + c) * 32 + lane];
    const v16h a1 = aL[(1 * NCHUNK + c) * 32 + lane];
    const v16h a2 = aL[(2 * NCHUNK + c) * 32 + lane];
    const v16h a3 = aL[(3 * NCHUNK + c) * 32 + lane];
    acc[0] = __builtin_amdgcn_wmma_f32_16x16x32_f16(false, a0, false, bf.v, (short)0, acc[0], false, false);
    acc[1] = __builtin_amdgcn_wmma_f32_16x16x32_f16(false, a1, false, bf.v, (short)0, acc[1], false, false);
    acc[2] = __builtin_amdgcn_wmma_f32_16x16x32_f16(false, a2, false, bf.v, (short)0, acc[2], false, false);
    acc[3] = __builtin_amdgcn_wmma_f32_16x16x32_f16(false, a3, false, bf.v, (short)0, acc[3], false, false);
  }

  // Store C (layout: lane -> N = lane&15, vgpr r -> M = r + 8*(lane>>4)),
  // and accumulate GN group partials: group(channel) = 2*t + (lane>>4).
  __shared__ float sb[8][4][2][2];     // [wave][t][laneGroup][sum,sumsq]
  const int hg   = h0 + wv;
  const int wcol = w0 + (lane & 15);
#pragma unroll
  for (int t = 0; t < 4; ++t) {
    float s = 0.f, q = 0.f;
#pragma unroll
    for (int r = 0; r < 8; ++r) {
      const float v = acc[t][r];
      const int o = t * 16 + r + ((lane >> 4) << 3);
      y[(((size_t)b * COUT + o) * HH + hg) * WW + wcol] = v;
      s += v; q += v * v;
    }
#pragma unroll
    for (int off = 1; off < 16; off <<= 1) {         // reduce within 16-lane half
      s += __shfl_xor(s, off, 32);
      q += __shfl_xor(q, off, 32);
    }
    if ((lane & 15) == 0) {
      sb[wv][t][lane >> 4][0] = s;
      sb[wv][t][lane >> 4][1] = q;
    }
  }
  __syncthreads();
  if (tid < 16) {
    const int g = tid >> 1, st = tid & 1;
    const int t = g >> 1, gh2 = g & 1;
    float s = 0.f;
    for (int w2 = 0; w2 < 8; ++w2) s += sb[w2][t][gh2][st];
    const int tile = thy * TILES_W + twx;
    statp[(((size_t)b * NTILES + tile) * NGRP + g) * 2 + st] = s;
  }
}

// ---------------------------------------------------------------------------
// K4: reduce 288 tile partials per (b, group) -> mean, rstd.
// ---------------------------------------------------------------------------
__global__ __launch_bounds__(128) void stats_kernel(const float* __restrict__ statp,
                                                    float* __restrict__ mr) {
  const int tid = threadIdx.x;
  if (tid >= NB * NGRP) return;
  const int b = tid >> 3, g = tid & 7;
  float s = 0.f, q = 0.f;
  for (int t = 0; t < NTILES; ++t) {
    const float* p = &statp[(((size_t)b * NTILES + t) * NGRP + g) * 2];
    s += p[0]; q += p[1];
  }
  const float invN = 1.0f / (float)((COUT / NGRP) * PLANE);   // 8*36864
  const float mean = s * invN;
  const float var  = q * invN - mean * mean;
  const float rstd = rsqrtf(var + EPS_GN);
  mr[(b * NGRP + g) * 2 + 0] = mean;
  mr[(b * NGRP + g) * 2 + 1] = rstd;
}

// ---------------------------------------------------------------------------
// K5: in-place GroupNorm + LeakyReLU on d_out, float4 vectorized.
// ---------------------------------------------------------------------------
__global__ __launch_bounds__(256) void gn_kernel(float* __restrict__ yio,
                                                 const float* __restrict__ mr,
                                                 const float* __restrict__ gamma,
                                                 const float* __restrict__ beta) {
  const size_t i4   = (size_t)blockIdx.x * 256 + threadIdx.x;
  const size_t base = i4 * 4;
  const int chan = (int)(base / PLANE);        // b*64 + o
  const int o = chan & 63, b = chan >> 6, g = o >> 3;
  const float mean = mr[(b * NGRP + g) * 2 + 0];
  const float rstd = mr[(b * NGRP + g) * 2 + 1];
  const float ga = gamma[o] * rstd;
  const float be = beta[o] - mean * ga;
  float4 v = *(const float4*)(yio + base);
  float r0 = v.x * ga + be; r0 = r0 >= 0.f ? r0 : NEG_SL * r0;
  float r1 = v.y * ga + be; r1 = r1 >= 0.f ? r1 : NEG_SL * r1;
  float r2 = v.z * ga + be; r2 = r2 >= 0.f ? r2 : NEG_SL * r2;
  float r3 = v.w * ga + be; r3 = r3 >= 0.f ? r3 : NEG_SL * r3;
  *(float4*)(yio + base) = make_float4(r0, r1, r2, r3);
}

// ---------------------------------------------------------------------------
extern "C" void kernel_launch(void* const* d_in, const int* in_sizes, int n_in,
                              void* d_out, int out_size, void* d_ws, size_t ws_size,
                              hipStream_t stream) {
  const float* x     = (const float*)d_in[0];
  const float* w1    = (const float*)d_in[1];
  const float* w2    = (const float*)d_in[2];
  const float* b2    = (const float*)d_in[3];
  const float* wts   = (const float*)d_in[4];
  const float* gamma = (const float*)d_in[5];
  const float* beta  = (const float*)d_in[6];
  float* out = (float*)d_out;

  // Workspace layout (256-byte aligned sections, ~1.51 MB total)
  char* ws = (char*)d_ws;
  float*    pool_part = (float*)(ws + 0);                  // 1024*9 f32   = 36864 B
  float*    attn      = (float*)(ws + 36864);              // 64 f32       = 256 B
  _Float16* afilt     = (_Float16*)(ws + 37120);           // 589824 h     = 1179648 B
  float*    statp     = (float*)(ws + 1216768);            // 16*288*8*2   = 294912 B
  float*    mr        = (float*)(ws + 1511680);            // 16*8*2 f32   = 1024 B

  pool_kernel<<<dim3(9, NB * CIN), 256, 0, stream>>>(x, pool_part);
  attn_kernel<<<1, 256, 0, stream>>>(pool_part, w1, w2, b2, attn);
  filt_kernel<<<(NB * 4 * NCHUNK * 32 * 16) / 256, 256, 0, stream>>>(attn, wts, afilt);
  conv_kernel<<<dim3(TILES_W, TILES_H, NB), 256, 0, stream>>>(x, afilt, out, statp);
  stats_kernel<<<1, 128, 0, stream>>>(statp, mr);
  gn_kernel<<<(NB * COUT * PLANE) / (4 * 256), 256, 0, stream>>>(out, mr, gamma, beta);
}
